// SelectiveCopyingModel_40467181863471
// MI455X (gfx1250) — compile-verified, matrix operand order checked by reading, not run on hardware
//
#include <hip/hip_runtime.h>
#include <hip/hip_bf16.h>
#include <stdint.h>

// Problem constants (match reference)
#define BB   32
#define SS   4096
#define VV   256
#define EE   128
#define HH   256
#define NMM  16
#define GG   512           // 2*H
#define NCH  32            // scan chunks per sequence
#define CL   (SS / NCH)    // 128 steps per chunk

typedef __bf16 bf16_t;
typedef bf16_t bf16x8  __attribute__((ext_vector_type(8)));
typedef bf16_t bf16x16 __attribute__((ext_vector_type(16)));
typedef float  f32x8   __attribute__((ext_vector_type(8)));
typedef unsigned int u32x4 __attribute__((ext_vector_type(4)));
typedef int i32x4 __attribute__((ext_vector_type(4)));
typedef int i32x8 __attribute__((ext_vector_type(8)));

// ---------------- f32 -> bf16 conversion (weights) ----------------
__global__ void cvt_f32_bf16_kernel(const float* __restrict__ in,
                                    bf16_t* __restrict__ out, int n) {
  int i = blockIdx.x * blockDim.x + threadIdx.x;
  int stride = gridDim.x * blockDim.x;
  for (; i < n; i += stride) out[i] = (bf16_t)in[i];
}

// ---------------- embedding gather -> bf16 activations ----------------
__global__ void embed_kernel(const int* __restrict__ x,
                             const float* __restrict__ emb,
                             bf16_t* __restrict__ out) {
  int bs = blockIdx.x;
  int t  = threadIdx.x;                 // 0..127
  int tok = x[bs];
  out[(size_t)bs * EE + t] = (bf16_t)emb[(size_t)tok * EE + t];
}

// ---------------- WMMA bf16 GEMM:  C[M,N] = A[M,K] * W[N,K]^T (+bias) ----------------
// Block: 256 threads = 8 waves; block tile 64(M) x 64(N); each wave 16(M) x 32(N).
// Weight tile (64 rows x K bf16) is DMA'd into LDS once per block by the Tensor
// Data Mover with end-of-row padding (row stride K+8 halves -> no ds bank camping),
// then all waves read B fragments from LDS while streaming A from global.
__device__ __forceinline__ bf16x16 load_frag_g(const bf16_t* __restrict__ p) {
  bf16x8 lo = *(const bf16x8*)(p);
  bf16x8 hi = *(const bf16x8*)(p + 16);
  return __builtin_shufflevector(lo, hi, 0,1,2,3,4,5,6,7,8,9,10,11,12,13,14,15);
}

__global__ __launch_bounds__(256)
void wmma_gemm_bias_kernel(const bf16_t* __restrict__ A,
                           const bf16_t* __restrict__ W,
                           const float* __restrict__ bias,
                           float* __restrict__ C,
                           int M, int N, int K) {
  __shared__ __align__(16) bf16_t sW[64 * (HH + 8)];   // worst case K=256: 33 KB

  const int tid  = threadIdx.x;
  const int lane = tid & 31;
  const int wave = tid >> 5;
  const int wm   = wave & 3;        // 4 waves along M
  const int wn   = wave >> 2;       // 2 waves along N
  const int rowBase = blockIdx.x * 64 + wm * 16;
  const int colBase = blockIdx.y * 64 + wn * 32;

  // ---- TDM: load this block's 64 x K weight tile into LDS (wave 0 issues) ----
  if (wave == 0) {
    const unsigned lds32 = (unsigned)(uintptr_t)(&sW[0]);
    const unsigned long long ga =
        (unsigned long long)(uintptr_t)(W + (size_t)blockIdx.y * 64 * K);
    const int kdw = K >> 1;                                  // row length in dwords
    const int pic = (31 - __builtin_clz((unsigned)kdw)) - 1; // pad_interval code
    // D# group0: count=1, lds_addr, global_addr[56:0], type=2
    u32x4 g0 = { 1u, lds32, (unsigned)ga,
                 (unsigned)((ga >> 32) & 0x1FFFFFFull) | (2u << 30) };
    // D# group1: data_size=2B, pad_enable, pad_interval=row, pad_amount=4 dwords,
    // tensor_dim0=K, tensor_dim1=64, tile_dim0=K, tile_dim1=64, dim0_stride=K
    i32x8 g1 = { (int)((1u << 16) | (1u << 20) | ((unsigned)pic << 22) | (3u << 25)),
                 (int)(((unsigned)K & 0xFFFFu) << 16),
                 (int)(((unsigned)K >> 16) | (64u << 16)),
                 (int)(((unsigned)K & 0xFFFFu) << 16),
                 (int)64,
                 (int)(unsigned)K,
                 0, 0 };
    i32x4 z4 = { 0, 0, 0, 0 };
    i32x8 z8 = { 0, 0, 0, 0, 0, 0, 0, 0 };
    __builtin_amdgcn_tensor_load_to_lds(g0, g1, z4, z4, z8, 0);
    __builtin_amdgcn_s_wait_tensorcnt((short)0);
  }
  __syncthreads();

  const int mrow = rowBase + (lane & 15);
  const int koff = (lane >> 4) * 8;
  const int lstride = K + 8;                              // padded LDS row stride
  const int sb0 = (wn * 32 + (lane & 15)) * lstride + koff;
  const int sb1 = sb0 + 16 * lstride;

  const bf16_t* pA = A + (size_t)mrow * K + koff;

  f32x8 acc0 = {0.f,0.f,0.f,0.f,0.f,0.f,0.f,0.f};
  f32x8 acc1 = {0.f,0.f,0.f,0.f,0.f,0.f,0.f,0.f};

  for (int kc = 0; kc < K; kc += 32) {
    __builtin_prefetch(pA + kc + 32, 0, 0);
    bf16x16 a = load_frag_g(pA + kc);
    bf16x8 l0 = *(const bf16x8*)(&sW[sb0 + kc]);
    bf16x8 h0 = *(const bf16x8*)(&sW[sb0 + kc + 16]);
    bf16x8 l1 = *(const bf16x8*)(&sW[sb1 + kc]);
    bf16x8 h1 = *(const bf16x8*)(&sW[sb1 + kc + 16]);
    bf16x16 b0 = __builtin_shufflevector(l0, h0, 0,1,2,3,4,5,6,7,8,9,10,11,12,13,14,15);
    bf16x16 b1 = __builtin_shufflevector(l1, h1, 0,1,2,3,4,5,6,7,8,9,10,11,12,13,14,15);
    acc0 = __builtin_amdgcn_wmma_f32_16x16x32_bf16(false, a, false, b0,
                                                   (short)0, acc0, false, false);
    acc1 = __builtin_amdgcn_wmma_f32_16x16x32_bf16(false, a, false, b1,
                                                   (short)0, acc1, false, false);
  }

  const int n0 = colBase + (lane & 15);
  const float bias0 = bias ? bias[n0]      : 0.0f;
  const float bias1 = bias ? bias[n0 + 16] : 0.0f;
  const int mBase = rowBase + (lane >> 4) * 8;
#pragma unroll
  for (int v = 0; v < 8; ++v) {
    const size_t m = (size_t)(mBase + v);
    C[m * N + n0]      = acc0[v] + bias0;
    C[m * N + n0 + 16] = acc1[v] + bias1;
  }
}

// ---------------- chunked parallel minGRU scan ----------------
// h_t = a_t h_{t-1} + b_t,  a = 1 - sigmoid(gate),  b = sigmoid(gate)*g(hid)
// Phase 1: per-(b,c,chunk) compose affine map (A,B) over CL steps.
// Phase 2: per-(b,c) 32-step scan over chunk maps -> chunk start states.
// Phase 3: replay each chunk from its start state, add residual, emit f32+bf16.
__global__ __launch_bounds__(256)
void scan_phase1_kernel(const float* __restrict__ gh,
                        float* __restrict__ Ach, float* __restrict__ Bch) {
  const int idx = blockIdx.x * blockDim.x + threadIdx.x;  // B*NCH*H
  const int c = idx & (HH - 1);
  const int r = idx >> 8;
  const int chunk = r & (NCH - 1);
  const int b = r >> 5;
  const float* g = gh + ((size_t)b * SS + (size_t)chunk * CL) * GG + c;
  float A = 1.0f, Bv = 0.0f;
  for (int s = 0; s < CL; ++s) {
    const float gate = g[(size_t)s * GG];
    const float hid  = g[(size_t)s * GG + HH];
    const float z  = 1.0f / (1.0f + __expf(-gate));
    const float gv = (hid >= 0.0f) ? (hid + 0.5f)
                                   : (1.0f / (1.0f + __expf(-hid)));
    const float a = 1.0f - z;
    A  = a * A;
    Bv = a * Bv + z * gv;
  }
  Ach[idx] = A;
  Bch[idx] = Bv;
}

__global__ __launch_bounds__(256)
void scan_phase2_kernel(const float* __restrict__ Ach,
                        const float* __restrict__ Bch,
                        float* __restrict__ Hst) {
  const int idx = blockIdx.x * blockDim.x + threadIdx.x;  // B*H
  const int c = idx & (HH - 1);
  const int b = idx >> 8;
  float h = 0.5f;                                         // h_0 = g(0)
  for (int k = 0; k < NCH; ++k) {
    const size_t o = ((size_t)(b * NCH + k) << 8) + c;
    Hst[o] = h;
    h = Ach[o] * h + Bch[o];
  }
}

__global__ __launch_bounds__(256)
void scan_phase3_kernel(const float* __restrict__ gh,
                        const float* __restrict__ Hst,
                        const float* res, float* hout,
                        bf16_t* __restrict__ hbf) {
  const int idx = blockIdx.x * blockDim.x + threadIdx.x;  // B*NCH*H
  const int c = idx & (HH - 1);
  const int r = idx >> 8;
  const int chunk = r & (NCH - 1);
  const int b = r >> 5;
  const float* g = gh + ((size_t)b * SS + (size_t)chunk * CL) * GG + c;
  float h = Hst[idx];
  for (int s = 0; s < CL; ++s) {
    const float gate = g[(size_t)s * GG];
    const float hid  = g[(size_t)s * GG + HH];
    const float z  = 1.0f / (1.0f + __expf(-gate));
    const float gv = (hid >= 0.0f) ? (hid + 0.5f)
                                   : (1.0f / (1.0f + __expf(-hid)));
    h += z * (gv - h);
    const size_t o = ((size_t)b * SS + (size_t)chunk * CL + s) * HH + c;
    const float outv = h + res[o];         // res/hout alias: same-thread RAW only
    hout[o] = outv;
    hbf[o]  = (bf16_t)outv;
  }
}

// ---------------- gather last NM positions into compact [B*NM, H] bf16 ----------------
__global__ void gather_last_kernel(const bf16_t* __restrict__ hbf,
                                   bf16_t* __restrict__ cmp) {
  const int row = blockIdx.x;        // b*NM + j
  const int c   = threadIdx.x;       // 0..255
  const int b   = row >> 4;
  const int j   = row & (NMM - 1);
  const int s   = SS - NMM + j;
  cmp[(size_t)row * HH + c] = hbf[((size_t)b * SS + s) * HH + c];
}

// ---------------- host-side launch ----------------
extern "C" void kernel_launch(void* const* d_in, const int* in_sizes, int n_in,
                              void* d_out, int out_size, void* d_ws, size_t ws_size,
                              hipStream_t stream) {
  const int*   x      = (const int*)  d_in[0];
  const float* emb    = (const float*)d_in[1];
  const float* W0     = (const float*)d_in[2];
  const float* b0     = (const float*)d_in[3];
  const float* Wl     = (const float*)d_in[4];
  const float* bl     = (const float*)d_in[5];
  const float* Walign = (const float*)d_in[6];
  const float* Wout   = (const float*)d_in[7];
  const float* bout   = (const float*)d_in[8];
  float* out = (float*)d_out;        // [B, NM, V] f32

  char* p = (char*)d_ws;
  auto carve = [&](size_t bytes) -> char* {
    char* r = p;
    p += (bytes + 255) & ~(size_t)255;
    return r;
  };
  bf16_t* W0b   = (bf16_t*)carve((size_t)GG * EE * 2);
  bf16_t* Wlb   = (bf16_t*)carve((size_t)2 * GG * HH * 2);
  bf16_t* Walb  = (bf16_t*)carve((size_t)HH * EE * 2);
  bf16_t* Woutb = (bf16_t*)carve((size_t)VV * HH * 2);
  bf16_t* hin   = (bf16_t*)carve((size_t)BB * SS * EE * 2);     // 32 MB
  float*  gh    = (float*) carve((size_t)BB * SS * GG * 4);     // 256 MB
  float*  hres  = (float*) carve((size_t)BB * SS * HH * 4);     // 128 MB
  bf16_t* hbf   = (bf16_t*)carve((size_t)BB * SS * HH * 2);     // 64 MB
  bf16_t* cmp   = (bf16_t*)carve((size_t)BB * NMM * HH * 2);
  float*  Ach   = (float*) carve((size_t)BB * NCH * HH * 4);    // 1 MB
  float*  Bch   = (float*) carve((size_t)BB * NCH * HH * 4);    // 1 MB
  float*  Hst   = (float*) carve((size_t)BB * NCH * HH * 4);    // 1 MB

  const size_t Mfull = (size_t)BB * SS;   // 131072 rows
  const int scan3_blocks = (BB * NCH * HH) / 256;   // 1024

  // 1) weights -> bf16
  {
    int n;
    n = GG * EE;      cvt_f32_bf16_kernel<<<(n + 255) / 256, 256, 0, stream>>>(W0,     W0b,   n);
    n = 2 * GG * HH;  cvt_f32_bf16_kernel<<<(n + 255) / 256, 256, 0, stream>>>(Wl,     Wlb,   n);
    n = HH * EE;      cvt_f32_bf16_kernel<<<(n + 255) / 256, 256, 0, stream>>>(Walign, Walb,  n);
    n = VV * HH;      cvt_f32_bf16_kernel<<<(n + 255) / 256, 256, 0, stream>>>(Wout,   Woutb, n);
  }

  // 2) embedding gather
  embed_kernel<<<(unsigned)Mfull, EE, 0, stream>>>(x, emb, hin);

  // 3) layer 0 GEMMs
  wmma_gemm_bias_kernel<<<dim3((unsigned)(Mfull / 64), GG / 64), 256, 0, stream>>>(
      hin, W0b, b0, gh, (int)Mfull, GG, EE);
  wmma_gemm_bias_kernel<<<dim3((unsigned)(Mfull / 64), HH / 64), 256, 0, stream>>>(
      hin, Walb, nullptr, hres, (int)Mfull, HH, EE);

  //    layer 0 scan
  scan_phase1_kernel<<<scan3_blocks, 256, 0, stream>>>(gh, Ach, Bch);
  scan_phase2_kernel<<<(BB * HH) / 256, 256, 0, stream>>>(Ach, Bch, Hst);
  scan_phase3_kernel<<<scan3_blocks, 256, 0, stream>>>(gh, Hst, hres, hres, hbf);

  // 4) layers 1..2
  for (int i = 0; i < 2; ++i) {
    wmma_gemm_bias_kernel<<<dim3((unsigned)(Mfull / 64), GG / 64), 256, 0, stream>>>(
        hbf, Wlb + (size_t)i * GG * HH, bl + (size_t)i * GG, gh, (int)Mfull, GG, HH);
    scan_phase1_kernel<<<scan3_blocks, 256, 0, stream>>>(gh, Ach, Bch);
    scan_phase2_kernel<<<(BB * HH) / 256, 256, 0, stream>>>(Ach, Bch, Hst);
    scan_phase3_kernel<<<scan3_blocks, 256, 0, stream>>>(gh, Hst, hres, hres, hbf);
  }

  // 5) output head on last NM positions
  gather_last_kernel<<<BB * NMM, HH, 0, stream>>>(hbf, cmp);
  wmma_gemm_bias_kernel<<<dim3((BB * NMM) / 64, VV / 64), 256, 0, stream>>>(
      cmp, Woutb, bout, out, BB * NMM, VV, HH);
}
